// ProsodyPredictor_11510512353361
// MI455X (gfx1250) — compile-verified, matrix operand order checked by reading, not run on hardware
//
#include <hip/hip_runtime.h>
#include <hip/hip_bf16.h>
#include <cstdint>

// ---------------- problem constants ----------------
constexpr int B  = 128;
constexpr int S  = 512;
constexpr int C  = 256;
constexpr int DG = 256;
constexpr int DO = 4;
constexpr int G  = 10;

constexpr int KX = 288;          // padded (C+DO)=260 -> 288 (9 k-tiles of 32)
constexpr int K0 = KX + DG;      // 544: combined [x | h0] K for layer 0
constexpr int K1 = DG + DG;      // 512: combined [h0 | h1] K for layer 1
constexpr int NH = 96;           // padded head outputs (10 + 40 + 40 -> 96)
constexpr int KH = DG + C;       // 512 : hcs = [H | enc]

constexpr float ZONEOUT = 0.1f;

// output layout (floats, concatenated in reference return order)
constexpr size_t O_OUTS = 0;
constexpr size_t O_PI   = O_OUTS + (size_t)B * S * DO;
constexpr size_t O_SIG  = O_PI   + (size_t)B * S * G;
constexpr size_t O_MU   = O_SIG  + (size_t)B * S * G * DO;
constexpr size_t O_MASK = O_MU   + (size_t)B * S * G * DO;

typedef __bf16 bf16;
typedef __attribute__((ext_vector_type(16))) __bf16 v16bf;
typedef __attribute__((ext_vector_type(8)))  __bf16 v8bf;
typedef __attribute__((ext_vector_type(8)))  float  v8f;

__device__ inline v8f v8f_zero() {
  v8f z = {0.f, 0.f, 0.f, 0.f, 0.f, 0.f, 0.f, 0.f};
  return z;
}

// Load one 16x32 bf16 fragment (A or B operand) per the CDNA5 16-bit layout:
// lane L supplies row/col (L&15); K runs [8*half, 8*half+7] and [16+8*half, +7].
// Both runs are 16-byte contiguous -> two b128 loads (global or ds).
__device__ inline v16bf load_frag(const bf16* __restrict__ row, int koff, int half) {
  union { v16bf v; v8bf h[2]; } u;
  u.h[0] = *reinterpret_cast<const v8bf*>(row + koff + 8 * half);
  u.h[1] = *reinterpret_cast<const v8bf*>(row + koff + 16 + 8 * half);
  return u.v;
}

__device__ inline v8f wmma_bf16(v16bf a, v16bf b, v8f c) {
  // D = A x B + C, fp32 accumulate
  return __builtin_amdgcn_wmma_f32_16x16x32_bf16(false, a, false, b, (short)0, c,
                                                 false, false);
}

__device__ inline float sigmoidf_dev(float x) { return 1.0f / (1.0f + __expf(-x)); }

// Sense-reversing device-wide barrier (16 resident blocks).
__device__ inline void grid_sync(unsigned* cnt, unsigned* gen, unsigned nblocks) {
  __threadfence();
  __syncthreads();
  if (threadIdx.x == 0) {
    unsigned g = __hip_atomic_load(gen, __ATOMIC_RELAXED, __HIP_MEMORY_SCOPE_AGENT);
    unsigned arrived =
        __hip_atomic_fetch_add(cnt, 1u, __ATOMIC_ACQ_REL, __HIP_MEMORY_SCOPE_AGENT);
    if (arrived == nblocks - 1) {
      __hip_atomic_store(cnt, 0u, __ATOMIC_RELAXED, __HIP_MEMORY_SCOPE_AGENT);
      __hip_atomic_fetch_add(gen, 1u, __ATOMIC_RELEASE, __HIP_MEMORY_SCOPE_AGENT);
    } else {
      while (__hip_atomic_load(gen, __ATOMIC_ACQUIRE, __HIP_MEMORY_SCOPE_AGENT) == g)
        __builtin_amdgcn_s_sleep(2);
    }
  }
  __syncthreads();
  __threadfence();
}

// ---------------- prep kernels ----------------
__global__ void init_state_kernel(float* h0f, float* h1f, bf16* h0b, bf16* h1b,
                                  unsigned* bar) {
  const int i = blockIdx.x * blockDim.x + threadIdx.x;
  const int n = 2 * B * DG;
  if (i < n) {
    h0f[i] = 0.f; h1f[i] = 0.f;
    h0b[i] = (bf16)0.f; h1b[i] = (bf16)0.f;
  }
  if (i == 0) { bar[0] = 0u; bar[64] = 0u; }
}

// wc0: (3*DG) x K0, row n = [Wih0[n,0:260] | 0-pad to 288 | Whh0[n,0:256]]
__global__ void build_wc0_kernel(const float* __restrict__ Wih0,
                                 const float* __restrict__ Whh0, bf16* __restrict__ wc0) {
  const int i = blockIdx.x * blockDim.x + threadIdx.x;
  if (i >= 3 * DG * K0) return;
  const int n = i / K0, k = i % K0;
  float v = 0.f;
  if (k < C + DO)   v = Wih0[(size_t)n * (C + DO) + k];
  else if (k >= KX) v = Whh0[(size_t)n * DG + (k - KX)];
  wc0[i] = (bf16)v;
}

// wc1: (3*DG) x K1, row n = [Wih1[n,:] | Whh1[n,:]]
__global__ void build_wc1_kernel(const float* __restrict__ Wih1,
                                 const float* __restrict__ Whh1, bf16* __restrict__ wc1) {
  const int i = blockIdx.x * blockDim.x + threadIdx.x;
  if (i >= 3 * DG * K1) return;
  const int n = i / K1, k = i % K1;
  const float v = (k < DG) ? Wih1[(size_t)n * DG + k] : Whh1[(size_t)n * DG + (k - DG)];
  wc1[i] = (bf16)v;
}

// wh: NH x KH, rows = [W_pi(10) | W_sigma(40) | W_mu(40) | zero-pad(6)]
__global__ void build_wh_kernel(const float* __restrict__ W_pi,
                                const float* __restrict__ W_sig,
                                const float* __restrict__ W_mu, bf16* __restrict__ wh) {
  const int i = blockIdx.x * blockDim.x + threadIdx.x;
  if (i >= NH * KH) return;
  const int r = i / KH, k = i % KH;
  float v = 0.f;
  if (r < G)                     v = W_pi[(size_t)r * KH + k];
  else if (r < G + G * DO)       v = W_sig[(size_t)(r - G) * KH + k];
  else if (r < G + 2 * G * DO)   v = W_mu[(size_t)(r - G - G * DO) * KH + k];
  wh[i] = (bf16)v;
}

// xs: (B*S) x KX bf16 = [enc(256) | prenet(4) | 0-pad(28)]
__global__ void build_xs_kernel(const float* __restrict__ enc, const float* __restrict__ tgt,
                                const float* __restrict__ W_pre,
                                const float* __restrict__ b_pre, bf16* __restrict__ xs) {
  const int i = blockIdx.x * blockDim.x + threadIdx.x;
  if (i >= B * S * KX) return;
  const int bs = i / KX, k = i % KX;
  const int s = bs % S;
  float v = 0.f;
  if (k < C) {
    v = enc[(size_t)bs * C + k];
  } else if (k < C + DO) {
    const int j = k - C;
    v = b_pre[j];
    if (s > 0) {  // go-frame at s==0 is zeros -> prenet = b_pre
      const float* prev = tgt + (size_t)(bs - 1) * DO;
      for (int d = 0; d < DO; ++d) v += prev[d] * W_pre[j * DO + d];
    }
  }
  xs[i] = (bf16)v;
}

// hcs columns [256,512) = enc in bf16 (columns [0,256) filled by the scan)
__global__ void build_hcs_enc_kernel(const float* __restrict__ enc, bf16* __restrict__ hcs) {
  const int i = blockIdx.x * blockDim.x + threadIdx.x;
  if (i >= B * S * C) return;
  const int bs = i / C, k = i % C;
  hcs[(size_t)bs * KH + DG + k] = (bf16)enc[i];
}

// ---------------- persistent recurrent kernel ----------------
// 16 blocks x 8 waves. nT = blockIdx (hidden tile), mT = local wave (batch tile).
// All 8 waves of a block share one nT -> their B operands (weight rows) are
// staged once in LDS (~101 KB of the 320 KB WGP LDS) and re-read per step with
// ds_load_b128 instead of global loads. One device-wide barrier per step
// (between layer0 and layer1); the end-of-step barrier is provably redundant
// with depth-2 ping-pong h buffers.
__global__ __launch_bounds__(256, 1) void gru_scan_kernel(
    const bf16* __restrict__ xs, const bf16* __restrict__ wc0, const bf16* __restrict__ wc1,
    const float* __restrict__ bih0, const float* __restrict__ bhh0,
    const float* __restrict__ bih1, const float* __restrict__ bhh1,
    float* h0f, float* h1f, bf16* h0b, bf16* h1b, bf16* __restrict__ hcs,
    unsigned* barcnt, unsigned* bargen) {
  __shared__ bf16 ws0[3 * 16 * K0];   // [gate][row][k], 52.2 KB
  __shared__ bf16 ws1[3 * 16 * K1];   // [gate][row][k], 49.2 KB

  const int lane  = threadIdx.x & 31;
  const int half  = lane >> 4;
  const int r15   = lane & 15;
  const int mT    = threadIdx.x >> 5;        // 0..7 : batch tile
  const int nT    = blockIdx.x;              // 0..15: hidden tile
  const int mBase = mT * 16;
  const int nCol  = nT * 16 + r15;           // this lane's hidden output column
  const int aRow  = mBase + r15;             // A-matrix row supplied by this lane
  const int HS    = B * DG;

  // stage this block's weight rows (B operands) into LDS once
  {
    v8bf* d0 = reinterpret_cast<v8bf*>(ws0);
    v8bf* d1 = reinterpret_cast<v8bf*>(ws1);
    const int R0 = K0 / 8;  // 68 v8bf chunks per layer-0 row
    const int R1 = K1 / 8;  // 64
    for (int u = threadIdx.x; u < 3 * 16 * R0; u += blockDim.x) {
      const int g = u / (16 * R0), rem = u % (16 * R0);
      const int j = rem / R0, kc = rem % R0;
      d0[u] = *reinterpret_cast<const v8bf*>(
          wc0 + (size_t)(g * DG + nT * 16 + j) * K0 + kc * 8);
    }
    for (int u = threadIdx.x; u < 3 * 16 * R1; u += blockDim.x) {
      const int g = u / (16 * R1), rem = u % (16 * R1);
      const int j = rem / R1, kc = rem % R1;
      d1[u] = *reinterpret_cast<const v8bf*>(
          wc1 + (size_t)(g * DG + nT * 16 + j) * K1 + kc * 8);
    }
  }
  __syncthreads();

  const float br0 = bih0[nCol] + bhh0[nCol];
  const float bz0 = bih0[DG + nCol] + bhh0[DG + nCol];
  const float bi0 = bih0[2 * DG + nCol];
  const float bh0 = bhh0[2 * DG + nCol];
  const float br1 = bih1[nCol] + bhh1[nCol];
  const float bz1 = bih1[DG + nCol] + bhh1[DG + nCol];
  const float bi1 = bih1[2 * DG + nCol];
  const float bh1 = bhh1[2 * DG + nCol];

  // per-lane B-operand row pointers into LDS (row index = r15 within the tile)
  const bf16* wR0 = ws0 + (size_t)(0 * 16 + r15) * K0;
  const bf16* wZ0 = ws0 + (size_t)(1 * 16 + r15) * K0;
  const bf16* wN0 = ws0 + (size_t)(2 * 16 + r15) * K0;
  const bf16* wR1 = ws1 + (size_t)(0 * 16 + r15) * K1;
  const bf16* wZ1 = ws1 + (size_t)(1 * 16 + r15) * K1;
  const bf16* wN1 = ws1 + (size_t)(2 * 16 + r15) * K1;

  const bf16* xrow_base = xs + (size_t)aRow * S * KX;

#pragma unroll 1
  for (int t = 0; t < S; ++t) {
    const int cur = t & 1, nxt = cur ^ 1;

    // ---------- layer 0 : gates from [x_t | h0] (K=544) ----------
    v8f aR = v8f_zero(), aZ = v8f_zero(), aNi = v8f_zero(), aNh = v8f_zero();
    {
      const bf16* xrow  = xrow_base + (size_t)t * KX;
      const bf16* h0row = h0b + (size_t)cur * HS + (size_t)aRow * DG;
      for (int kt = 0; kt < K0 / 32; ++kt) {
        const int ko = kt * 32;
        v16bf a = (ko < KX) ? load_frag(xrow, ko, half) : load_frag(h0row, ko - KX, half);
        aR = wmma_bf16(a, load_frag(wR0, ko, half), aR);
        aZ = wmma_bf16(a, load_frag(wZ0, ko, half), aZ);
        v16bf bn = load_frag(wN0, ko, half);
        if (ko < KX) aNi = wmma_bf16(a, bn, aNi);   // inn (x part of n-gate)
        else         aNh = wmma_bf16(a, bn, aNh);   // hn  (h part of n-gate)
      }
      const float* hc  = h0f + (size_t)cur * HS;
      float*       hnx = h0f + (size_t)nxt * HS;
      bf16*        hb  = h0b + (size_t)nxt * HS;
#pragma unroll
      for (int i = 0; i < 8; ++i) {
        const int   row  = mBase + i + 8 * half;
        const float rv   = sigmoidf_dev(aR[i] + br0);
        const float zv   = sigmoidf_dev(aZ[i] + bz0);
        const float nv   = tanhf(aNi[i] + bi0 + rv * (aNh[i] + bh0));
        const float hp   = hc[(size_t)row * DG + nCol];
        const float hnew = (1.f - zv) * nv + zv * hp;
        const float ho   = ZONEOUT * hp + (1.f - ZONEOUT) * hnew;
        hnx[(size_t)row * DG + nCol] = ho;
        hb[(size_t)row * DG + nCol]  = (bf16)ho;
      }
    }
    grid_sync(barcnt, bargen, gridDim.x);  // h0' visible to all blocks

    // ---------- layer 1 : gates from [h0' | h1] (K=512) ----------
    aR = v8f_zero(); aZ = v8f_zero(); aNi = v8f_zero(); aNh = v8f_zero();
    {
      const bf16* h0n = h0b + (size_t)nxt * HS + (size_t)aRow * DG;
      const bf16* h1c = h1b + (size_t)cur * HS + (size_t)aRow * DG;
      for (int kt = 0; kt < K1 / 32; ++kt) {
        const int ko = kt * 32;
        v16bf a = (ko < DG) ? load_frag(h0n, ko, half) : load_frag(h1c, ko - DG, half);
        aR = wmma_bf16(a, load_frag(wR1, ko, half), aR);
        aZ = wmma_bf16(a, load_frag(wZ1, ko, half), aZ);
        v16bf bn = load_frag(wN1, ko, half);
        if (ko < DG) aNi = wmma_bf16(a, bn, aNi);
        else         aNh = wmma_bf16(a, bn, aNh);
      }
      const float* hc  = h1f + (size_t)cur * HS;
      float*       hnx = h1f + (size_t)nxt * HS;
      bf16*        hb  = h1b + (size_t)nxt * HS;
#pragma unroll
      for (int i = 0; i < 8; ++i) {
        const int   row  = mBase + i + 8 * half;
        const float rv   = sigmoidf_dev(aR[i] + br1);
        const float zv   = sigmoidf_dev(aZ[i] + bz1);
        const float nv   = tanhf(aNi[i] + bi1 + rv * (aNh[i] + bh1));
        const float hp   = hc[(size_t)row * DG + nCol];
        const float hnew = (1.f - zv) * nv + zv * hp;
        const float ho   = ZONEOUT * hp + (1.f - ZONEOUT) * hnew;
        hnx[(size_t)row * DG + nCol] = ho;
        hb[(size_t)row * DG + nCol]  = (bf16)ho;
        hcs[((size_t)row * S + t) * KH + nCol] = (bf16)ho;  // H half of hcs
      }
    }
    // no end-of-step barrier needed: next step's layer0 only writes h0[cur],
    // whose last readers (layer0 of this step) are behind the mid-step barrier,
    // and next step's layer1 reads are behind the next mid-step barrier.
  }
}

// ---------------- heads GEMM: P(65536 x 96) = hcs(65536 x 512) @ wh^T ----------------
__global__ __launch_bounds__(256) void heads_gemm_kernel(const bf16* __restrict__ hcs,
                                                         const bf16* __restrict__ wh,
                                                         float* __restrict__ P) {
  const int lane = threadIdx.x & 31;
  const int half = lane >> 4;
  const int r15  = lane & 15;
  const int wave = blockIdx.x * (blockDim.x >> 5) + (threadIdx.x >> 5);
  const int mT   = wave / (NH / 16);
  const int nT   = wave % (NH / 16);
  const bf16* arow = hcs + ((size_t)mT * 16 + r15) * KH;
  const bf16* brow = wh + ((size_t)nT * 16 + r15) * KH;
  v8f acc = v8f_zero();
  for (int kt = 0; kt < KH / 32; ++kt) {
    const int ko = kt * 32;
    acc = wmma_bf16(load_frag(arow, ko, half), load_frag(brow, ko, half), acc);
  }
#pragma unroll
  for (int i = 0; i < 8; ++i)
    P[((size_t)mT * 16 + i + 8 * half) * NH + nT * 16 + r15] = acc[i];
}

// ---------------- finalize: outs copy, softmax(pi), elu(sigma)+1, mu ----------------
__global__ void finalize_kernel(const float* __restrict__ P, const float* __restrict__ tgt,
                                const float* __restrict__ b_pi, const float* __restrict__ b_sig,
                                const float* __restrict__ b_mu, float* __restrict__ out) {
  const int idx = blockIdx.x * blockDim.x + threadIdx.x;
  if (idx >= B * S) return;
  const float* p = P + (size_t)idx * NH;
  for (int d = 0; d < DO; ++d)
    out[O_OUTS + (size_t)idx * DO + d] = tgt[(size_t)idx * DO + d];
  float l[G], m = -3.4e38f;
  for (int g = 0; g < G; ++g) { l[g] = p[g] + b_pi[g]; m = fmaxf(m, l[g]); }
  float ssum = 0.f;
  for (int g = 0; g < G; ++g) { l[g] = __expf(l[g] - m); ssum += l[g]; }
  const float inv = 1.f / ssum;
  for (int g = 0; g < G; ++g) out[O_PI + (size_t)idx * G + g] = l[g] * inv;
  for (int e = 0; e < G * DO; ++e) {
    const float x = p[G + e] + b_sig[e];
    out[O_SIG + (size_t)idx * G * DO + e] = (x > 0.f) ? (x + 1.f) : __expf(x);
  }
  for (int e = 0; e < G * DO; ++e)
    out[O_MU + (size_t)idx * G * DO + e] = p[G + G * DO + e] + b_mu[e];
}

__global__ void mask_kernel(const int* __restrict__ dur, float* __restrict__ out) {
  const int b = threadIdx.x;
  if (b >= B) return;
  int cnt = 0;
  for (int s = 0; s < S; ++s) cnt += (dur[(size_t)b * S + s] > 0) ? 1 : 0;
  for (int s = 0; s < S; ++s)
    out[O_MASK + (size_t)b * S + s] = (s >= cnt) ? 1.0f : 0.0f;
}

// ---------------- host launcher ----------------
extern "C" void kernel_launch(void* const* d_in, const int* in_sizes, int n_in,
                              void* d_out, int out_size, void* d_ws, size_t ws_size,
                              hipStream_t stream) {
  (void)in_sizes; (void)n_in; (void)out_size; (void)ws_size;
  const float* enc   = (const float*)d_in[0];
  const float* tgt   = (const float*)d_in[1];
  const int*   dur   = (const int*)d_in[2];
  const float* W_pre = (const float*)d_in[3];
  const float* b_pre = (const float*)d_in[4];
  const float* Wih0  = (const float*)d_in[5];
  const float* Whh0  = (const float*)d_in[6];
  const float* bih0  = (const float*)d_in[7];
  const float* bhh0  = (const float*)d_in[8];
  const float* Wih1  = (const float*)d_in[9];
  const float* Whh1  = (const float*)d_in[10];
  const float* bih1  = (const float*)d_in[11];
  const float* bhh1  = (const float*)d_in[12];
  const float* W_pi  = (const float*)d_in[13];
  const float* b_pi  = (const float*)d_in[14];
  const float* W_sig = (const float*)d_in[15];
  const float* b_sig = (const float*)d_in[16];
  const float* W_mu  = (const float*)d_in[17];
  const float* b_mu  = (const float*)d_in[18];
  float* out = (float*)d_out;

  char* ws = (char*)d_ws;
  size_t off = 0;
  auto alloc = [&](size_t bytes) {
    size_t o = off;
    off = (off + bytes + 255) & ~(size_t)255;
    return o;
  };
  unsigned* bar = (unsigned*)(ws + alloc(256));                // [0]=count, [64]=gen
  bf16*  xs  = (bf16*)(ws + alloc((size_t)B * S * KX * 2));    // 37.7 MB
  bf16*  hcs = (bf16*)(ws + alloc((size_t)B * S * KH * 2));    // 67.1 MB
  bf16*  wc0 = (bf16*)(ws + alloc((size_t)3 * DG * K0 * 2));
  bf16*  wc1 = (bf16*)(ws + alloc((size_t)3 * DG * K1 * 2));
  bf16*  wh  = (bf16*)(ws + alloc((size_t)NH * KH * 2));
  float* h0f = (float*)(ws + alloc((size_t)2 * B * DG * 4));
  float* h1f = (float*)(ws + alloc((size_t)2 * B * DG * 4));
  bf16*  h0b = (bf16*)(ws + alloc((size_t)2 * B * DG * 2));
  bf16*  h1b = (bf16*)(ws + alloc((size_t)2 * B * DG * 2));
  float* P   = (float*)(ws + alloc((size_t)B * S * NH * 4));   // 25.2 MB

  init_state_kernel<<<(2 * B * DG + 255) / 256, 256, 0, stream>>>(h0f, h1f, h0b, h1b, bar);
  build_wc0_kernel<<<(3 * DG * K0 + 255) / 256, 256, 0, stream>>>(Wih0, Whh0, wc0);
  build_wc1_kernel<<<(3 * DG * K1 + 255) / 256, 256, 0, stream>>>(Wih1, Whh1, wc1);
  build_wh_kernel<<<(NH * KH + 255) / 256, 256, 0, stream>>>(W_pi, W_sig, W_mu, wh);
  build_xs_kernel<<<(B * S * KX + 255) / 256, 256, 0, stream>>>(enc, tgt, W_pre, b_pre, xs);
  build_hcs_enc_kernel<<<(B * S * C + 255) / 256, 256, 0, stream>>>(enc, hcs);

  gru_scan_kernel<<<16, 256, 0, stream>>>(xs, wc0, wc1, bih0, bhh0, bih1, bhh1,
                                          h0f, h1f, h0b, h1b, hcs, bar, bar + 64);

  heads_gemm_kernel<<<(B * S / 16) * (NH / 16) / 8, 256, 0, stream>>>(hcs, wh, P);
  finalize_kernel<<<(B * S + 255) / 256, 256, 0, stream>>>(P, tgt, b_pi, b_sig, b_mu, out);
  mask_kernel<<<1, 128, 0, stream>>>(dur, out);
}
